// DeformableBlock_4217657885306
// MI455X (gfx1250) — matrix operand order repaired and checked by reading, not compiled, not run on hardware
//
#include <hip/hip_runtime.h>
#include <math.h>

// ---------------------------------------------------------------------------
// DeformableBlock, MI455X (gfx1250, wave32, WMMA).
// B=8, L=4, P=1024, C=256, NH=8, NS=4, HD=32; levels 64^2,32^2,16^2,8^2.
// Strategy: pre-embed features (feat^T @ embed_w) with bf16 WMMA, then
// projections (x @ [w_attn|w_off]) with bf16 WMMA, then per-(row,head)
// bilinear sampling of the 32-channel pre-embedded maps.
// ---------------------------------------------------------------------------

typedef __attribute__((ext_vector_type(8)))  float  v8f;
typedef __attribute__((ext_vector_type(16))) __bf16 v16bf;

#define BB 8
#define LL 4
#define PP 1024
#define CC 256
#define NHH 8
#define NSS 4
#define HDD 32
#define SUMHW 5440  // 4096+1024+256+64

static __device__ __forceinline__ unsigned short f2bf(float f) {
  unsigned int u = __float_as_uint(f);
  unsigned int r = u + 0x7fffu + ((u >> 16) & 1u);  // round-to-nearest-even
  return (unsigned short)(r >> 16);
}
static __device__ __forceinline__ unsigned int pk2(float a, float b) {
  return (unsigned int)f2bf(a) | ((unsigned int)f2bf(b) << 16);
}

union AB16 {
  v16bf v;
  unsigned int u[8];
  uint4 q[2];
};

// B-fragment: 16 contiguous bf16 (K = base..base+15 for fixed column) from LDS.
static __device__ __forceinline__ v16bf ld_bfrag(const unsigned short* p) {
  AB16 t;
  t.q[0] = *(const uint4*)(p);
  t.q[1] = *(const uint4*)(p + 8);
  return t.v;
}

// ---------------------------------------------------------------------------
// Kernel A: femb[b, lvl, y, x, d] = sum_c feat[b,c,y,x] * embed_w[lvl][c,d]
// One 4-wave workgroup per (b, 64-pixel tile). Each wave owns a 16px M-tile.
// ---------------------------------------------------------------------------
__global__ __launch_bounds__(128) void preembed_kernel(
    const float* __restrict__ feat,    // (B, 256, H, W)
    const float* __restrict__ embw,    // (256, 32) for this level
    float* __restrict__ femb,          // (B, SUMHW, 32)
    int HW, int lvloff) {
  __shared__ unsigned short lW[HDD * CC];   // [d][c] bf16, 16 KB
  __shared__ unsigned short stg[64 * 32];   // [px][c] bf16, 4 KB

  const int tilesPerB = HW >> 6;
  const int b    = blockIdx.x / tilesPerB;
  const int tile = blockIdx.x % tilesPerB;
  const int pix0 = tile * 64;
  const int tid  = threadIdx.x;
  const int lane = tid & 31;
  const int wave = tid >> 5;

  // Transpose embed_w into LDS as [d][c] (one time).
  for (int i = tid; i < CC * HDD; i += 128) {
    int c = i >> 5, d = i & 31;
    lW[d * CC + c] = f2bf(embw[i]);  // embw[c*32+d] == embw[i]
  }
  __syncthreads();

  v8f acc0 = {}, acc1 = {};
  const float* fb = feat + (size_t)b * CC * HW;
  const int ch = tid & 31, pxg = tid >> 5;
  const int m     = lane & 15;
  const int base0 = (lane >> 4) * 8;    // A-frag K split per lane half
  const int baseB = (lane >> 4) * 16;   // B-frag K base per lane half

  for (int kb = 0; kb < 8; ++kb) {
    // Stage 32 channels x 64 pixels (coalesced: pixels contiguous per channel).
    const float* src = fb + (size_t)(kb * 32 + ch) * HW + pix0 + pxg * 16;
    float4 f0 = *(const float4*)(src + 0);
    float4 f1 = *(const float4*)(src + 4);
    float4 f2 = *(const float4*)(src + 8);
    float4 f3 = *(const float4*)(src + 12);
    float vv[16] = {f0.x, f0.y, f0.z, f0.w, f1.x, f1.y, f1.z, f1.w,
                    f2.x, f2.y, f2.z, f2.w, f3.x, f3.y, f3.z, f3.w};
#pragma unroll
    for (int j = 0; j < 16; ++j)
      stg[(pxg * 16 + j) * 32 + ch] = f2bf(vv[j]);
    __syncthreads();

    // A fragment: lane half selects contiguous K run {16h..16h+15}, matching B.
    const unsigned short* arow = &stg[(wave * 16 + m) * 32];
    AB16 a;
    a.q[0] = *(const uint4*)(arow + 2 * base0);
    a.q[1] = *(const uint4*)(arow + 2 * base0 + 8);

    // B fragments: column d = lane&15 (+16), K contiguous in [d][c] LDS.
    v16bf b0 = ld_bfrag(&lW[(lane & 15) * CC + kb * 32 + baseB]);
    v16bf b1 = ld_bfrag(&lW[((lane & 15) + 16) * CC + kb * 32 + baseB]);

    acc0 = __builtin_amdgcn_wmma_f32_16x16x32_bf16(false, a.v, false, b0,
                                                   (short)0, acc0, false, false);
    acc1 = __builtin_amdgcn_wmma_f32_16x16x32_bf16(false, a.v, false, b1,
                                                   (short)0, acc1, false, false);
    __syncthreads();
  }

  // C/D layout: col n = lane&15, row m = j + 8*(lane>=16).
  float* outb = femb + ((size_t)(b * SUMHW + lvloff + pix0 + wave * 16)) * HDD;
  const int n  = lane & 15;
  const int mo = 8 * (lane >> 4);
#pragma unroll
  for (int j = 0; j < 8; ++j) {
    outb[(mo + j) * HDD + n]      = acc0[j];
    outb[(mo + j) * HDD + 16 + n] = acc1[j];
  }
}

// ---------------------------------------------------------------------------
// Kernel B1: proj[r, 0:32] = x[r] @ w_attn ; proj[r, 32:96] = x[r] @ w_off
// 4 waves/WG, each wave does 4 sequential 16-row M-tiles (256 rows per WG).
// ---------------------------------------------------------------------------
__global__ __launch_bounds__(128) void proj_kernel(
    const float* __restrict__ x,       // (32768, 256)
    const float* __restrict__ w_attn,  // (256, 32)
    const float* __restrict__ w_off,   // (256, 64)
    float* __restrict__ proj) {        // (32768, 96)
  __shared__ unsigned short lW[96 * CC];  // [n][c] bf16, 48 KB

  const int tid  = threadIdx.x;
  const int lane = tid & 31;
  const int wave = tid >> 5;

  for (int i = tid; i < 96 * CC; i += 128) {
    int c = i / 96, n = i % 96;
    float v = (n < 32) ? w_attn[c * 32 + n] : w_off[c * 64 + (n - 32)];
    lW[n * CC + c] = f2bf(v);
  }
  __syncthreads();

  const int rowBlock = blockIdx.x * 256;
  const int m     = lane & 15;
  const int base0 = (lane >> 4) * 8;
  const int baseB = (lane >> 4) * 16;

  for (int it = 0; it < 4; ++it) {
    const int rowBase = rowBlock + it * 64 + wave * 16;
    const float* xr = x + (size_t)(rowBase + m) * CC;

    v8f acc[6];
#pragma unroll
    for (int nt = 0; nt < 6; ++nt) acc[nt] = (v8f){};

    for (int kb = 0; kb < 8; ++kb) {
      // A fragment straight from global: one contiguous 16-float run (K = 16h..16h+15).
      const float* p0 = xr + kb * 32 + 2 * base0;
      const float* p1 = xr + kb * 32 + 2 * base0 + 8;
      float4 f0 = *(const float4*)(p0);
      float4 f1 = *(const float4*)(p0 + 4);
      float4 f2 = *(const float4*)(p1);
      float4 f3 = *(const float4*)(p1 + 4);
      AB16 a;
      a.u[0] = pk2(f0.x, f0.y); a.u[1] = pk2(f0.z, f0.w);
      a.u[2] = pk2(f1.x, f1.y); a.u[3] = pk2(f1.z, f1.w);
      a.u[4] = pk2(f2.x, f2.y); a.u[5] = pk2(f2.z, f2.w);
      a.u[6] = pk2(f3.x, f3.y); a.u[7] = pk2(f3.z, f3.w);

#pragma unroll
      for (int nt = 0; nt < 6; ++nt) {
        v16bf bb = ld_bfrag(&lW[(nt * 16 + (lane & 15)) * CC + kb * 32 + baseB]);
        acc[nt] = __builtin_amdgcn_wmma_f32_16x16x32_bf16(
            false, a.v, false, bb, (short)0, acc[nt], false, false);
      }
    }

    const int n  = lane & 15;
    const int mo = 8 * (lane >> 4);
#pragma unroll
    for (int nt = 0; nt < 6; ++nt)
#pragma unroll
      for (int j = 0; j < 8; ++j)
        proj[(size_t)(rowBase + mo + j) * 96 + nt * 16 + n] = acc[nt][j];
  }
}

// ---------------------------------------------------------------------------
// Kernel B2: per (row, head): softmax over NS, tanh offsets, bilinear sample
// of the 32-channel pre-embedded level map, weighted sum, +embed_b.
// ---------------------------------------------------------------------------
__global__ __launch_bounds__(256) void sample_kernel(
    const float* __restrict__ ref,      // (B, L, P, 2)
    const float* __restrict__ b_attn,   // (32,)
    const float* __restrict__ b_off,    // (64,)
    const float* __restrict__ embed_b,  // (L, 32)
    const float* __restrict__ femb,     // (B, SUMHW, 32)
    const float* __restrict__ proj,     // (32768, 96)
    float* __restrict__ out) {          // (B, L, P, 256)
  const int gid = blockIdx.x * 256 + threadIdx.x;
  if (gid >= BB * LL * PP * NHH) return;
  const int head = gid & 7;
  const int r    = gid >> 3;         // ((b*L + l)*P + p)
  const int l    = (r >> 10) & 3;
  const int b    = r >> 12;

  const int Hs[4]   = {64, 32, 16, 8};
  const int offs[4] = {0, 4096, 5120, 5376};
  const int H = Hs[l], W = Hs[l];
  const float* fb = femb + ((size_t)(b * SUMHW + offs[l])) * HDD;
  const float* pr = proj + (size_t)r * 96;

  // softmax over NS=4 samples of this head
  float lg[4], w[4];
#pragma unroll
  for (int s = 0; s < 4; ++s) lg[s] = pr[head * 4 + s] + b_attn[head * 4 + s];
  float mx = fmaxf(fmaxf(lg[0], lg[1]), fmaxf(lg[2], lg[3]));
  float sum = 0.f;
#pragma unroll
  for (int s = 0; s < 4; ++s) { w[s] = __expf(lg[s] - mx); sum += w[s]; }
  const float inv = 1.f / sum;
#pragma unroll
  for (int s = 0; s < 4; ++s) w[s] *= inv;

  const float rx = ref[(size_t)r * 2 + 0];
  const float ry = ref[(size_t)r * 2 + 1];

  float4 acc[8];
#pragma unroll
  for (int k = 0; k < 8; ++k) acc[k] = make_float4(0.f, 0.f, 0.f, 0.f);

#pragma unroll
  for (int s = 0; s < 4; ++s) {
    const int j = head * 4 + s;
    const float ox = tanhf(pr[32 + j * 2 + 0] + b_off[j * 2 + 0]);
    const float oy = tanhf(pr[32 + j * 2 + 1] + b_off[j * 2 + 1]);
    const float fx = (rx + ox + 1.f) * 0.5f * (float)(W - 1);
    const float fy = (ry + oy + 1.f) * 0.5f * (float)(H - 1);
    const float x0f = floorf(fx), y0f = floorf(fy);
    const int x0 = (int)x0f, y0 = (int)y0f;
    const float wx1 = fx - x0f, wy1 = fy - y0f;
    const float wx0 = 1.f - wx1, wy0 = 1.f - wy1;
    const float ws = w[s];

    const int   ys[2] = {y0, y0 + 1};
    const int   xs[2] = {x0, x0 + 1};
    const float wy[2] = {wy0, wy1};
    const float wx[2] = {wx0, wx1};
#pragma unroll
    for (int cy = 0; cy < 2; ++cy) {
#pragma unroll
      for (int cx = 0; cx < 2; ++cx) {
        const int yi = ys[cy], xi = xs[cx];
        if (yi >= 0 && yi < H && xi >= 0 && xi < W) {
          const float cw = ws * wy[cy] * wx[cx];
          const float4* c = (const float4*)(fb + ((size_t)yi * W + xi) * HDD);
#pragma unroll
          for (int k = 0; k < 8; ++k) {
            float4 v = c[k];
            acc[k].x += cw * v.x; acc[k].y += cw * v.y;
            acc[k].z += cw * v.z; acc[k].w += cw * v.w;
          }
        }
      }
    }
  }

  // out[r, head*32 + d] = acc[d] + embed_b[l, d]  (softmax weights sum to 1)
  float* o = out + (size_t)r * (NHH * HDD) + head * HDD;
  const float* eb = embed_b + l * HDD;
#pragma unroll
  for (int k = 0; k < 8; ++k) {
    float4 v = acc[k];
    v.x += eb[4 * k + 0]; v.y += eb[4 * k + 1];
    v.z += eb[4 * k + 2]; v.w += eb[4 * k + 3];
    *(float4*)(o + 4 * k) = v;
  }
}

// ---------------------------------------------------------------------------
extern "C" void kernel_launch(void* const* d_in, const int* in_sizes, int n_in,
                              void* d_out, int out_size, void* d_ws, size_t ws_size,
                              hipStream_t stream) {
  const float* x       = (const float*)d_in[0];
  const float* ref     = (const float*)d_in[1];
  const float* feats[4] = {(const float*)d_in[2], (const float*)d_in[3],
                           (const float*)d_in[4], (const float*)d_in[5]};
  const float* w_attn  = (const float*)d_in[6];
  const float* b_attn  = (const float*)d_in[7];
  const float* w_off   = (const float*)d_in[8];
  const float* b_off   = (const float*)d_in[9];
  const float* embed_w = (const float*)d_in[10];
  const float* embed_b = (const float*)d_in[11];
  float* out = (float*)d_out;

  float* ws   = (float*)d_ws;
  float* femb = ws;                                   // B*SUMHW*32 = 1,392,640 f32
  float* proj = ws + (size_t)BB * SUMHW * HDD;        // 32768*96   = 3,145,728 f32

  const int HWs[4]  = {4096, 1024, 256, 64};
  const int offs[4] = {0, 4096, 5120, 5376};

  // A: pre-embed each level (feat^T @ embed_w), bf16 WMMA.
  for (int lvl = 0; lvl < 4; ++lvl) {
    const int tiles = HWs[lvl] / 64;
    preembed_kernel<<<dim3(BB * tiles), dim3(128), 0, stream>>>(
        feats[lvl], embed_w + (size_t)lvl * CC * HDD, femb, HWs[lvl], offs[lvl]);
  }

  // B1: projections x @ [w_attn | w_off], bf16 WMMA. 32768 rows / 256 per WG.
  proj_kernel<<<dim3(128), dim3(128), 0, stream>>>(x, w_attn, w_off, proj);

  // B2: softmax + tanh + bilinear gather + combine. One thread per (row, head).
  const int tasks = BB * LL * PP * NHH;  // 262144
  sample_kernel<<<dim3(tasks / 256), dim3(256), 0, stream>>>(
      ref, b_attn, b_off, embed_b, femb, proj, out);
}
